// PMPGNN_14353780703439
// MI455X (gfx1250) — matrix operand order repaired and checked by reading, not compiled, not run on hardware
//
#include <hip/hip_runtime.h>
#include <math.h>

#define HID   128
#define HEADS 4
#define FEAT  512     // HEADS*HID
#define CATW  1024
#define NEG_GAT 0.2f
#define NEG     0.01f
#define ALPHA_MIX 0.01f

typedef __attribute__((ext_vector_type(16))) _Float16 v16h;
typedef __attribute__((ext_vector_type(8)))  _Float16 v8h;
typedef __attribute__((ext_vector_type(4)))  _Float16 v4h;
typedef __attribute__((ext_vector_type(2)))  _Float16 h2v;
typedef __attribute__((ext_vector_type(8)))  float    v8f;

#define BS_STRIDE 72   // halves; 144 B row stride: 16B-aligned, low bank conflicts

__device__ __forceinline__ float lk(float v, float s) { return v > 0.0f ? v : v * s; }

__device__ __forceinline__ h2v pk2(float a, float b) {
#if __has_builtin(__builtin_amdgcn_cvt_pkrtz)
    return __builtin_bit_cast(h2v, __builtin_amdgcn_cvt_pkrtz(a, b)); // v_cvt_pk_rtz_f16_f32
#else
    h2v r; r.x = (_Float16)a; r.y = (_Float16)b; return r;
#endif
}

__device__ __forceinline__ void atomicMaxF(float* addr, float val) {
    int* ia = (int*)addr;
    int old = __float_as_int(*addr);
    while (__int_as_float(old) < val) {
        int assumed = old;
        old = atomicCAS(ia, assumed, __float_as_int(val));
        if (old == assumed) break;
    }
}

// ---------------------------------------------------------------------------
// f32 -> f16 conversion pre-pass (n must be a multiple of 4; all sizes here are)
// ---------------------------------------------------------------------------
__global__ void k_f32_to_f16(const float* __restrict__ in, _Float16* __restrict__ out,
                             long long n) {
    long long i = (blockIdx.x * (long long)blockDim.x + threadIdx.x) * 4;
    if (i < n) {
        float4 f = *(const float4*)(in + i);
        union { v4h v; h2v p[2]; } u;
        u.p[0] = pk2(f.x, f.y);
        u.p[1] = pk2(f.z, f.w);
        *(v4h*)(out + i) = u.v;
    }
}

// ---------------------------------------------------------------------------
// Fast WMMA GEMM (requires Nn % 64 == 0, K % 32 == 0):
//   C[M,Nn] = leaky(A16 @ B + bias, slope),  A16 pre-converted to f16.
// 256-thread block = 8 waves -> 128x64 tile. B strip (32x64 f32) is software-
// pipelined through registers: next strip's global loads issue before the
// current step's 4 WMMAs, hiding global latency; cvt+ds_store at loop top.
// A fragments are two aligned 16-byte f16 loads (no VALU converts).
// No early returns: barriers stay uniform, EXEC all-ones at every WMMA.
// ---------------------------------------------------------------------------
__global__ void __launch_bounds__(256)
k_gemm_fast(const _Float16* __restrict__ A16, const float* __restrict__ B,
            const float* __restrict__ bias, float* __restrict__ C,
            int M, int Nn, int K, float slope, int mblocks) {
    __shared__ _Float16 Bs[32 * BS_STRIDE];

    const int mb  = blockIdx.x % mblocks;
    const int nb  = blockIdx.x / mblocks;
    const int bn0 = nb * 64;

    const int t     = threadIdx.x;
    const int widx  = t >> 5;
    const int l     = t & 31;
    const int lrow  = l & 15;
    const int half8 = (l >> 4) * 8;

    const int m0 = mb * 128 + widx * 16;
    int arowi = m0 + lrow; if (arowi >= M) arowi = M - 1;     // clamp (partial M blk)
    const _Float16* arow = A16 + (size_t)arowi * K;

    // cooperative B-stage mapping: 32 rows x 64 cols, 8 f32 per thread
    const int rr = t >> 3;
    const int c0 = (t & 7) * 8;
    const float* bgbase = B + (size_t)rr * Nn + bn0 + c0;
    _Float16* bsdst = Bs + rr * BS_STRIDE + c0;

    v8f acc0 = {}, acc1 = {}, acc2 = {}, acc3 = {};

    // pre-load first B strip into registers
    float4 f0 = ((const float4*)bgbase)[0];
    float4 f1 = ((const float4*)bgbase)[1];

    for (int k = 0; k < K; k += 32) {
        // convert staged strip and publish to LDS
        union { v8h v; h2v p[4]; } u;
        u.p[0] = pk2(f0.x, f0.y); u.p[1] = pk2(f0.z, f0.w);
        u.p[2] = pk2(f1.x, f1.y); u.p[3] = pk2(f1.z, f1.w);
        __syncthreads();                         // prior-iteration LDS reads done
        *(v8h*)bsdst = u.v;
        __syncthreads();

        // issue next strip's global loads now; waited at next loop top
        if (k + 32 < K) {
            const float4* g = (const float4*)(bgbase + (size_t)(k + 32) * Nn);
            f0 = g[0]; f1 = g[1];
        }
        if (k + 64 < K) __builtin_prefetch(arow + k + 64, 0, 1);  // global_prefetch_b8

        // A fragment: CDNA5 16-bit A 16x32 layout == two contiguous 8-half runs
        union { v16h v; v8h h[2]; } ua;
        ua.h[0] = *(const v8h*)(arow + k + half8);
        ua.h[1] = *(const v8h*)(arow + k + 16 + half8);

        // B fragments from LDS: lane l holds N-col l%16, K in same order as A
        union { v16h v; v8h h[2]; } b0, b1, b2, b3;
        #pragma unroll
        for (int s = 0; s < 16; ++s) {
            const _Float16* bp = Bs + (s < 8 ? half8 + s : 16 + half8 + s - 8) * BS_STRIDE + lrow;
            b0.v[s] = bp[0]; b1.v[s] = bp[16]; b2.v[s] = bp[32]; b3.v[s] = bp[48];
        }

        acc0 = __builtin_amdgcn_wmma_f32_16x16x32_f16(false, ua.v, false, b0.v, (short)0, acc0, false, false);
        acc1 = __builtin_amdgcn_wmma_f32_16x16x32_f16(false, ua.v, false, b1.v, (short)0, acc1, false, false);
        acc2 = __builtin_amdgcn_wmma_f32_16x16x32_f16(false, ua.v, false, b2.v, (short)0, acc2, false, false);
        acc3 = __builtin_amdgcn_wmma_f32_16x16x32_f16(false, ua.v, false, b3.v, (short)0, acc3, false, false);
    }

    // Store: D layout -> VGPR r: M = r + 8*(l>>4), N = lane%16
    v8f accs[4] = {acc0, acc1, acc2, acc3};
    #pragma unroll
    for (int sub = 0; sub < 4; ++sub) {
        int col = bn0 + sub * 16 + lrow;
        float bb = bias[col];
        #pragma unroll
        for (int r = 0; r < 8; ++r) {
            int row = m0 + r + half8;
            if (row < M)
                C[(size_t)row * Nn + col] = lk(accs[sub][r] + bb, slope);
        }
    }
}

// ---------------------------------------------------------------------------
// Generic guarded WMMA GEMM (one wave per 16x64 tile) — only for the final
// 20000x40x64 classifier GEMM where Nn is not a multiple of 64.
// ---------------------------------------------------------------------------
__global__ void k_gemm_any(const _Float16* __restrict__ A16, const float* __restrict__ B,
                           const float* __restrict__ bias, float* __restrict__ C,
                           int M, int Nn, int K, float slope, int mtiles, int ntiles) {
    int wid = blockIdx.x * (blockDim.x >> 5) + (threadIdx.x >> 5);
    if (wid >= mtiles * ntiles) return;
    int l     = threadIdx.x & 31;
    int lrow  = l & 15;
    int half8 = (l >> 4) * 8;
    int mt = wid % mtiles, nt = wid / mtiles;
    int m0 = mt * 16, n0 = nt * 64;

    int arowi = m0 + lrow; if (arowi >= M) arowi = M - 1;
    const _Float16* arow = A16 + (size_t)arowi * K;

    v8f acc[4] = {v8f{}, v8f{}, v8f{}, v8f{}};
    for (int k = 0; k < K; k += 32) {
        union { v16h v; v8h h[2]; } ua;
        ua.h[0] = *(const v8h*)(arow + k + half8);
        ua.h[1] = *(const v8h*)(arow + k + 16 + half8);
        #pragma unroll
        for (int sub = 0; sub < 4; ++sub) {
            int nbase = n0 + sub * 16;
            if (nbase < Nn) {
                v16h bv;
                #pragma unroll
                for (int e = 0; e < 16; ++e) {
                    int col = nbase + lrow;
                    int kk = k + (e < 8 ? half8 + e : 16 + half8 + e - 8);
                    bv[e] = (_Float16)((col < Nn) ? B[(size_t)kk * Nn + col] : 0.0f);
                }
                acc[sub] = __builtin_amdgcn_wmma_f32_16x16x32_f16(
                    false, ua.v, false, bv, (short)0, acc[sub], false, false);
            }
        }
    }
    #pragma unroll
    for (int sub = 0; sub < 4; ++sub) {
        int col = n0 + sub * 16 + lrow;
        if (col < Nn) {
            float bb = bias[col];
            #pragma unroll
            for (int r = 0; r < 8; ++r) {
                int row = m0 + r + half8;
                if (row < M)
                    C[(size_t)row * Nn + col] = lk(acc[sub][r] + bb, slope);
            }
        }
    }
}

// ---------------------------------------------------------------------------
// GATv2 edge kernels (segment softmax over dst with appended self-loops)
// ---------------------------------------------------------------------------
__device__ __forceinline__ void edge_sd(const int* __restrict__ src,
                                        const int* __restrict__ dst,
                                        int E, int e, int& s, int& d) {
    if (e < E) { s = src[e]; d = dst[e]; }
    else       { s = e - E; d = e - E; }          // self loops
}

__global__ void k_init_md(float* __restrict__ m, float* __restrict__ den, int n) {
    int i = blockIdx.x * blockDim.x + threadIdx.x;
    if (i < n) { m[i] = -1e30f; den[i] = 0.0f; }
}

__global__ void k_init_bias(float* __restrict__ out, const float* __restrict__ bias,
                            long long total, int w) {
    long long i = blockIdx.x * (long long)blockDim.x + threadIdx.x;
    if (i < total) out[i] = bias[(int)(i % w)];
}

__global__ void k_edge_logits(const float* __restrict__ xl, const float* __restrict__ xr,
                              const float* __restrict__ att,
                              const int* __restrict__ src, const int* __restrict__ dst,
                              int E, int Etot, float* __restrict__ logits,
                              float* __restrict__ m) {
    int i = blockIdx.x * blockDim.x + threadIdx.x;
    if (i >= Etot * HEADS) return;
    int e = i >> 2, h = i & 3;
    int s, d; edge_sd(src, dst, E, e, s, d);
    const float4* pl = (const float4*)(xl + (size_t)s * FEAT + h * HID);
    const float4* pr = (const float4*)(xr + (size_t)d * FEAT + h * HID);
    const float4* pa = (const float4*)(att + h * HID);
    float acc = 0.0f;
    #pragma unroll 4
    for (int c = 0; c < HID / 4; ++c) {
        float4 a = pa[c], u = pl[c], v = pr[c];
        acc = fmaf(a.x, lk(u.x + v.x, NEG_GAT), acc);
        acc = fmaf(a.y, lk(u.y + v.y, NEG_GAT), acc);
        acc = fmaf(a.z, lk(u.z + v.z, NEG_GAT), acc);
        acc = fmaf(a.w, lk(u.w + v.w, NEG_GAT), acc);
    }
    logits[i] = acc;
    atomicMaxF(&m[d * HEADS + h], acc);
}

__global__ void k_edge_z(const int* __restrict__ src, const int* __restrict__ dst,
                         int E, int Etot, float* __restrict__ logits,
                         const float* __restrict__ m, float* __restrict__ den) {
    int i = blockIdx.x * blockDim.x + threadIdx.x;
    if (i >= Etot * HEADS) return;
    int e = i >> 2, h = i & 3;
    int s, d; edge_sd(src, dst, E, e, s, d);
    float z = expf(logits[i] - m[d * HEADS + h]);
    logits[i] = z;
    atomicAdd(&den[d * HEADS + h], z);
}

__global__ void k_edge_scatter(const float* __restrict__ xl,
                               const int* __restrict__ src, const int* __restrict__ dst,
                               int E, int Etot, const float* __restrict__ z,
                               const float* __restrict__ den, float* __restrict__ out) {
    long long i = blockIdx.x * (long long)blockDim.x + threadIdx.x;
    if (i >= (long long)Etot * FEAT) return;
    int e = (int)(i / FEAT);
    int j = (int)(i % FEAT);
    int h = j >> 7;
    int s, d; edge_sd(src, dst, E, e, s, d);
    float coef = z[e * HEADS + h] / den[d * HEADS + h];
    atomicAdd(&out[(size_t)d * FEAT + j], xl[(size_t)s * FEAT + j] * coef);
}

// ---------------------------------------------------------------------------
// LayerNorm over the 1024-wide concat [(1-a)*x1, a*x2] (+optional residual),
// then leaky(0.01). One wave per node, 32 values per lane.
// ---------------------------------------------------------------------------
__global__ void k_ln(const float* __restrict__ x1, const float* __restrict__ x2,
                     const float* __restrict__ res, const float* __restrict__ g,
                     const float* __restrict__ b, float* __restrict__ out, int n_nodes) {
    int wid  = blockIdx.x * (blockDim.x >> 5) + (threadIdx.x >> 5);
    int lane = threadIdx.x & 31;
    if (wid >= n_nodes) return;
    const float* r1 = x1 + (size_t)wid * FEAT;
    const float* r2 = x2 + (size_t)wid * FEAT;
    float vals[32];
    float s = 0.0f;
    #pragma unroll
    for (int t = 0; t < 32; ++t) {
        int j = t * 32 + lane;
        float v = (j < FEAT) ? (1.0f - ALPHA_MIX) * r1[j] : ALPHA_MIX * r2[j - FEAT];
        if (res) v += res[(size_t)wid * CATW + j];
        vals[t] = v; s += v;
    }
    #pragma unroll
    for (int off = 16; off > 0; off >>= 1) s += __shfl_xor(s, off, 32);
    float mu = s * (1.0f / CATW);
    float vs = 0.0f;
    #pragma unroll
    for (int t = 0; t < 32; ++t) { float d0 = vals[t] - mu; vs += d0 * d0; }
    #pragma unroll
    for (int off = 16; off > 0; off >>= 1) vs += __shfl_xor(vs, off, 32);
    float inv = rsqrtf(vs * (1.0f / CATW) + 1e-5f);
    #pragma unroll
    for (int t = 0; t < 32; ++t) {
        int j = t * 32 + lane;
        float y = (vals[t] - mu) * inv * g[j] + b[j];
        out[(size_t)wid * CATW + j] = lk(y, NEG);
    }
}

__global__ void k_logsoftmax(float* __restrict__ out, int n, int w) {
    int i = blockIdx.x * blockDim.x + threadIdx.x;
    if (i >= n) return;
    float* row = out + (size_t)i * w;
    float mx = -1e30f;
    for (int j = 0; j < w; ++j) mx = fmaxf(mx, row[j]);
    float s = 0.0f;
    for (int j = 0; j < w; ++j) s += expf(row[j] - mx);
    float ls = logf(s) + mx;
    for (int j = 0; j < w; ++j) row[j] = row[j] - ls;
}

// ---------------------------------------------------------------------------
extern "C" void kernel_launch(void* const* d_in, const int* in_sizes, int n_in,
                              void* d_out, int out_size, void* d_ws, size_t ws_size,
                              hipStream_t stream) {
    (void)n_in; (void)ws_size; (void)out_size;
    const float* x   = (const float*)d_in[0];
    const int*   ei1 = (const int*)d_in[1];
    const int*   ei2 = (const int*)d_in[2];
    const int E1 = in_sizes[1] / 2;
    const int E2 = in_sizes[2] / 2;
    const int* src1 = ei1;       const int* dst1 = ei1 + E1;
    const int* src2 = ei2;       const int* dst2 = ei2 + E2;
    const int N = in_sizes[0] / 512;   // IN_DIM = 512

    const float* proj_w = (const float*)d_in[3];
    const float* proj_b = (const float*)d_in[4];
    // GAT param blocks: {Wl, bl, Wr, br, att, bias} at base indices 5,11,17,23
    const float* norm_g = (const float*)d_in[29];
    const float* norm_b = (const float*)d_in[30];
    const float* mlp_w1 = (const float*)d_in[31];
    const float* mlp_b1 = (const float*)d_in[32];
    const float* mlp_w2 = (const float*)d_in[33];
    const float* mlp_b2 = (const float*)d_in[34];

    // ---- workspace layout ----
    char* wsb = (char*)d_ws;
    size_t off = 0;
    auto alloc = [&](size_t bytes) { size_t p = off; off += (bytes + 255) & ~(size_t)255; return p; };
    float* h    = (float*)(wsb + alloc((size_t)N * HID  * 4));
    float* xl   = (float*)(wsb + alloc((size_t)N * FEAT * 4));
    float* xr   = (float*)(wsb + alloc((size_t)N * FEAT * 4));
    float* x1   = (float*)(wsb + alloc((size_t)N * FEAT * 4));
    float* x2   = (float*)(wsb + alloc((size_t)N * FEAT * 4));
    float* h1   = (float*)(wsb + alloc((size_t)N * CATW * 4));
    float* h2   = (float*)(wsb + alloc((size_t)N * CATW * 4));
    float* mid  = (float*)(wsb + alloc((size_t)N * 64   * 4));
    float* mbuf = (float*)(wsb + alloc((size_t)N * HEADS * 4));
    float* dbuf = (float*)(wsb + alloc((size_t)N * HEADS * 4));
    int Emax = ((E1 > E2) ? E1 : E2) + N;
    float* zbuf = (float*)(wsb + alloc((size_t)Emax * HEADS * 4));
    // f16 activation buffers with disjoint lifetimes:
    //   a16s: x16 (proj) -> h16 (block1) -> mid16 (mlp2)
    //   a16b: h1_16 (block2) -> h2_16 (mlp1)
    _Float16* a16s = (_Float16*)(wsb + alloc((size_t)N * FEAT * 2));
    _Float16* a16b = (_Float16*)(wsb + alloc((size_t)N * CATW * 2));

    auto cvt16 = [&](const float* in, _Float16* out, long long n) {
        k_f32_to_f16<<<(int)((n / 4 + 255) / 256), 256, 0, stream>>>(in, out, n);
    };

    auto gemm = [&](const _Float16* A16, const float* B, const float* bias, float* C,
                    int M, int Nn, int K, float slope) {
        if ((Nn & 63) == 0) {
            int mblocks = (M + 127) / 128;
            int blocks  = mblocks * (Nn / 64);
            k_gemm_fast<<<blocks, 256, 0, stream>>>(A16, B, bias, C, M, Nn, K, slope, mblocks);
        } else {
            int mtiles = (M + 15) / 16;
            int ntiles = (Nn + 63) / 64;
            int blocks = (mtiles * ntiles + 7) / 8;
            k_gemm_any<<<blocks, 256, 0, stream>>>(A16, B, bias, C, M, Nn, K, slope, mtiles, ntiles);
        }
    };

    auto run_conv = [&](const _Float16* hin16, int K, const int* src, const int* dst,
                        int E, int pbase, float* outbuf) {
        const float* Wl   = (const float*)d_in[pbase + 0];
        const float* bl   = (const float*)d_in[pbase + 1];
        const float* Wr   = (const float*)d_in[pbase + 2];
        const float* br   = (const float*)d_in[pbase + 3];
        const float* att  = (const float*)d_in[pbase + 4];
        const float* bias = (const float*)d_in[pbase + 5];
        gemm(hin16, Wl, bl, xl, N, FEAT, K, 1.0f);   // slope=1 -> identity
        gemm(hin16, Wr, br, xr, N, FEAT, K, 1.0f);
        int Etot = E + N;
        k_init_md<<<(N * HEADS + 255) / 256, 256, 0, stream>>>(mbuf, dbuf, N * HEADS);
        long long totb = (long long)N * FEAT;
        k_init_bias<<<(int)((totb + 255) / 256), 256, 0, stream>>>(outbuf, bias, totb, FEAT);
        int eh = Etot * HEADS;
        k_edge_logits<<<(eh + 255) / 256, 256, 0, stream>>>(xl, xr, att, src, dst, E, Etot, zbuf, mbuf);
        k_edge_z<<<(eh + 255) / 256, 256, 0, stream>>>(src, dst, E, Etot, zbuf, mbuf, dbuf);
        long long tot = (long long)Etot * FEAT;
        k_edge_scatter<<<(int)((tot + 255) / 256), 256, 0, stream>>>(xl, src, dst, E, Etot, zbuf, dbuf, outbuf);
    };

    // 1) projection: h = leaky(x @ proj_w + proj_b, 0.01)
    cvt16(x, a16s, (long long)N * 512);
    gemm(a16s, proj_w, proj_b, h, N, HID, 512, NEG);

    // 2) block 1 (input h, K=128)
    cvt16(h, a16s, (long long)N * HID);
    run_conv(a16s, HID, src1, dst1, E1, 5,  x1);
    run_conv(a16s, HID, src2, dst2, E2, 11, x2);
    k_ln<<<(N + 7) / 8, 256, 0, stream>>>(x1, x2, nullptr, norm_g, norm_b, h1, N);

    // 3) block 2 (input h1, K=1024; residual in LN)
    cvt16(h1, a16b, (long long)N * CATW);
    run_conv(a16b, CATW, src1, dst1, E1, 17, x1);
    run_conv(a16b, CATW, src2, dst2, E2, 23, x2);
    k_ln<<<(N + 7) / 8, 256, 0, stream>>>(x1, x2, h1, norm_g, norm_b, h2, N);

    // 4) MLP head + log_softmax into d_out
    cvt16(h2, a16b, (long long)N * CATW);
    gemm(a16b, mlp_w1, mlp_b1, mid, N, 64, CATW, NEG);
    cvt16(mid, a16s, (long long)N * 64);
    gemm(a16s, mlp_w2, mlp_b2, (float*)d_out, N, 40, 64, 1.0f);
    k_logsoftmax<<<(N + 255) / 256, 256, 0, stream>>>((float*)d_out, N, 40);
}
